// MultiStreamSparseAutoencoder_83940840833383
// MI455X (gfx1250) — matrix operand order repaired and checked by reading, not compiled, not run on hardware
//
#include <hip/hip_runtime.h>
#include <hip/hip_bf16.h>
#include <cstdint>

// ---------------------------------------------------------------------------
// Types for CDNA5 WMMA (wave32): v16bf A/B fragments, v8f accumulator.
// ---------------------------------------------------------------------------
typedef __bf16 bf16;
typedef __bf16 v16bf __attribute__((ext_vector_type(16)));
typedef __bf16 v8bf  __attribute__((ext_vector_type(8)));
typedef __bf16 v2bf  __attribute__((ext_vector_type(2)));
typedef float  v8f   __attribute__((ext_vector_type(8)));

// TDM descriptor vector types (Tensor Data Mover, CDNA5 ISA)
typedef unsigned int u32x4 __attribute__((ext_vector_type(4)));
typedef int          i32x4 __attribute__((ext_vector_type(4)));
typedef int          i32x8 __attribute__((ext_vector_type(8)));

#define B_DIM 4096
#define L_DIM 32768
#define K_TOP 32

#define BM 128
#define BN 128
#define BK 32

// ---------------------------------------------------------------------------
// Encoder GEMM: logits[m,n] = sum_k (x[m,k]-pre_b[k]) * W_enc[n,k] + lat_b[n]
// bf16 WMMA 16x16x32, f32 accumulate. 256 threads = 8 waves, each wave owns a
// 64x32 sub-tile (4x2 grid of 16x16 accumulators) of the 128x128 block tile.
// Double-buffered LDS staging: one barrier per K-step, global loads for the
// next tile issued before the current tile's WMMAs.
// ---------------------------------------------------------------------------
__global__ __launch_bounds__(256, 2)
void sae_encode_gemm(const float* __restrict__ X,
                     const float* __restrict__ W,
                     const float* __restrict__ pre_b,
                     const float* __restrict__ lat_b,
                     float* __restrict__ logits,
                     int d /* inner dim: 768 or 1024 */)
{
    __shared__ bf16 As[2][BM][BK];   // (x - pre_b) tiles, 16 KB
    __shared__ bf16 Bs[2][BN][BK];   // W_enc tiles,       16 KB

    const int tid  = threadIdx.x;
    const int lane = tid & 31;
    const int wave = tid >> 5;        // 0..7
    const int wrow = wave >> 2;       // 0..1 -> M offset 64*wrow
    const int wcol = wave & 3;        // 0..3 -> N offset 32*wcol
    const int half = lane >> 4;       // 0: lanes 0-15, 1: lanes 16-31
    const int lrow = lane & 15;

    const int m0 = blockIdx.y * BM;
    const int n0 = blockIdx.x * BN;

    v8f acc[4][2];
#pragma unroll
    for (int i = 0; i < 4; ++i)
#pragma unroll
        for (int j = 0; j < 2; ++j)
            acc[i][j] = (v8f)(0.0f);

    // Per-thread staging slots: 8 element-pairs per tile for A and B each.
    float2 xa[8], wb[8], pb[8];

    auto stage_load = [&](int k0) {
#pragma unroll
        for (int j = 0; j < 8; ++j) {
            const int i = j * 256 + tid;  // 0..2047
            const int r = i >> 4;         // row 0..127
            const int c = i & 15;         // pair col 0..15
            xa[j] = *(const float2*)(X + (size_t)(m0 + r) * d + k0 + 2 * c);
            wb[j] = *(const float2*)(W + (size_t)(n0 + r) * d + k0 + 2 * c);
            pb[j] = *(const float2*)(pre_b + k0 + 2 * c);
        }
    };
    auto stage_store = [&](int buf) {
#pragma unroll
        for (int j = 0; j < 8; ++j) {
            const int i = j * 256 + tid;
            const int r = i >> 4;
            const int c = i & 15;
            v2bf p; p[0] = (bf16)(xa[j].x - pb[j].x); p[1] = (bf16)(xa[j].y - pb[j].y);
            *(v2bf*)&As[buf][r][2 * c] = p;
            v2bf q; q[0] = (bf16)wb[j].x; q[1] = (bf16)wb[j].y;
            *(v2bf*)&Bs[buf][r][2 * c] = q;
        }
    };

    const int nkt = d / BK;
    stage_load(0);
    stage_store(0);

    for (int kt = 0; kt < nkt; ++kt) {
        __syncthreads();                 // buf `cur` ready; buf `1-cur` free
        const int cur = kt & 1;

        if (kt + 1 < nkt) {
            stage_load((kt + 1) * BK);   // issue next-tile global loads early
            if (kt + 2 < nkt) {
                __builtin_prefetch(X + (size_t)(m0 + (tid >> 1)) * d + (kt + 2) * BK, 0, 1);
                __builtin_prefetch(W + (size_t)(n0 + (tid >> 1)) * d + (kt + 2) * BK, 0, 1);
            }
        }

        // A fragments (contiguous K order k = 16*half + s): lane holds row M=lrow;
        // lanes 0-15 take K=0..15, lanes 16-31 take K=16..31 (matches B).
        v16bf afrag[4];
#pragma unroll
        for (int i = 0; i < 4; ++i) {
            const bf16* ap = &As[cur][wrow * 64 + i * 16 + lrow][half * 16];
            v8bf lo = *(const v8bf*)ap;         // K = half*16 .. half*16+7
            v8bf hi = *(const v8bf*)(ap + 8);   // K = half*16+8 .. +7
            afrag[i] = __builtin_shufflevector(lo, hi,
                0, 1, 2, 3, 4, 5, 6, 7, 8, 9, 10, 11, 12, 13, 14, 15);
        }
        // B fragments (32x16): lane holds col N=lrow; lanes 0-15 K=0..15,
        // lanes 16-31 K=16..31 -> one contiguous 32-byte LDS read.
        v16bf bfrag[2];
#pragma unroll
        for (int j = 0; j < 2; ++j)
            bfrag[j] = *(const v16bf*)&Bs[cur][wcol * 32 + j * 16 + lrow][half * 16];

#pragma unroll
        for (int i = 0; i < 4; ++i)
#pragma unroll
            for (int j = 0; j < 2; ++j)
                acc[i][j] = __builtin_amdgcn_wmma_f32_16x16x32_bf16(
                    /*neg_a=*/false, afrag[i],
                    /*neg_b=*/false, bfrag[j],
                    /*c_mod=*/(short)0, acc[i][j],
                    /*reuse_a=*/false, /*reuse_b=*/false);

        if (kt + 1 < nkt)
            stage_store(1 - cur);        // fill the other buffer for kt+1
    }

    // Epilogue: C/D layout -> lane col N=lrow, VGPR v row M = v + half*8.
#pragma unroll
    for (int j = 0; j < 2; ++j) {
        const int n = n0 + wcol * 32 + j * 16 + lrow;
        const float lb = lat_b[n];
#pragma unroll
        for (int i = 0; i < 4; ++i) {
            const int mbase = m0 + wrow * 64 + i * 16 + half * 8;
#pragma unroll
            for (int v = 0; v < 8; ++v)
                logits[(size_t)(mbase + v) * L_DIM + n] = acc[i][j][v] + lb;
        }
    }
}

// ---------------------------------------------------------------------------
// Top-K gate: per-row top-32 of 32768 logits (descending, smaller index wins
// ties, matching lax.top_k), ReLU, scatter into zeroed codes + mask rows.
// Row staged in 128 KB LDS via the Tensor Data Mover (one 1x32768 f32 tile),
// waited with s_wait_tensorcnt.
// ---------------------------------------------------------------------------
__global__ __launch_bounds__(256)
void sae_topk_gate(const float* __restrict__ logits,
                   float* __restrict__ codes,
                   float* __restrict__ mask,
                   float* __restrict__ idx_out)
{
    __shared__ float buf[L_DIM];      // 128 KB (gfx1250: up to 320 KB per WG)
    __shared__ float rv[256];
    __shared__ int   ri[256];
    __shared__ float topv[K_TOP];
    __shared__ int   topi[K_TOP];

    const int row = blockIdx.x;
    const int tid = threadIdx.x;
    const float* src = logits + (size_t)row * L_DIM;

    // --- TDM: DMA the whole row global -> LDS (CDNA5 ISA)
    if (tid == 0) {
        const unsigned long long ga = (unsigned long long)(uintptr_t)src;
        const unsigned lds_addr = (unsigned)(uintptr_t)&buf[0];  // low 32b = LDS offset
        u32x4 g0;
        g0[0] = 0x1u;                                   // count=1, user descriptor
        g0[1] = lds_addr;                               // lds_addr [63:32]
        g0[2] = (unsigned)(ga & 0xffffffffu);           // global_addr [95:64]
        g0[3] = (unsigned)((ga >> 32) & 0x01ffffffu)    // global_addr [120:96]
              | (2u << 30);                             // type=2 ("image")
        i32x8 g1;
        g1[0] = 0x00020000;                             // data_size=2 (4B), no pad/iter
        g1[1] = (int)0x80000000;                        // tensor_dim0=32768 (lo16 @[31:16])
        g1[2] = 0x00010000;                             // dim0 hi16=0, tensor_dim1=1
        g1[3] = 0x00008000;                             // dim1 hi16=0, tile_dim0=32768
        g1[4] = 0x00000001;                             // tile_dim1=1, tile_dim2=0
        g1[5] = 0x00008000;                             // tensor_dim0_stride=32768 (lo32)
        g1[6] = (int)0x80000000;                        // stride0 hi16=0, stride1 lo16
        g1[7] = 0;                                      // stride1 hi32=0
        i32x4 z4 = (i32x4)0;
#if defined(__clang_major__) && (__clang_major__ >= 23)
        i32x8 z8 = (i32x8)0;
        __builtin_amdgcn_tensor_load_to_lds(g0, g1, z4, z4, z8, 0);
#else
        __builtin_amdgcn_tensor_load_to_lds(g0, g1, z4, z4, 0);
#endif
        __builtin_amdgcn_s_wait_tensorcnt(0);
    }
    __syncthreads();   // wave 0 passed its tensorcnt wait -> row is in LDS

    for (int it = 0; it < K_TOP; ++it) {
        float bv = -INFINITY;
        int   bi = L_DIM;
        for (int i = tid; i < L_DIM; i += 256) {
            const float v = buf[i];
            if (v > bv) { bv = v; bi = i; }
        }
        rv[tid] = bv; ri[tid] = bi;
        __syncthreads();
        for (int s = 128; s > 0; s >>= 1) {
            if (tid < s) {
                const float ov = rv[tid + s];
                const int   oi = ri[tid + s];
                if (ov > rv[tid] || (ov == rv[tid] && oi < ri[tid])) {
                    rv[tid] = ov; ri[tid] = oi;
                }
            }
            __syncthreads();
        }
        if (tid == 0) {
            topv[it] = rv[0];
            topi[it] = ri[0];
            buf[ri[0]] = -INFINITY;   // remove winner
        }
        __syncthreads();
    }

    // Zero-fill codes & mask rows, then scatter the K winners.
    float* crow = codes + (size_t)row * L_DIM;
    float* mrow = mask  + (size_t)row * L_DIM;
    const float4 z = make_float4(0.f, 0.f, 0.f, 0.f);
    for (int i = tid * 4; i < L_DIM; i += 256 * 4) {
        *(float4*)(crow + i) = z;
        *(float4*)(mrow + i) = z;
    }
    __threadfence();
    __syncthreads();
    if (tid < K_TOP) {
        const int   id = topi[tid];
        const float v  = topv[tid];
        idx_out[(size_t)row * K_TOP + tid] = (float)id;
        crow[id] = v > 0.0f ? v : 0.0f;   // ReLU post-activation
        mrow[id] = 1.0f;
    }
}

// ---------------------------------------------------------------------------
// Sparse decode: out[row,:] = pre_b + sum_k codes[row,idx_k] * W_dec[:,idx_k]
// W_dec is [d, L]; only K=32 columns touched per row (L2-resident gathers).
// ---------------------------------------------------------------------------
__global__ __launch_bounds__(256)
void sae_decode(const float* __restrict__ codes,
                const float* __restrict__ idxf,
                const float* __restrict__ Wdec,
                const float* __restrict__ pre_b,
                float* __restrict__ out,
                int d)
{
    __shared__ int   sidx[K_TOP];
    __shared__ float sval[K_TOP];
    const int row = blockIdx.x;
    const int tid = threadIdx.x;

    if (tid < K_TOP) {
        const int id = (int)idxf[(size_t)row * K_TOP + tid];
        sidx[tid] = id;
        sval[tid] = codes[(size_t)row * L_DIM + id];
    }
    __syncthreads();

    for (int e = tid; e < d; e += 256) {
        float acc = pre_b[e];
        const float* wrow = Wdec + (size_t)e * L_DIM;
#pragma unroll 8
        for (int k = 0; k < K_TOP; ++k)
            acc += sval[k] * wrow[sidx[k]];
        out[(size_t)row * d + e] = acc;
    }
}

// Mask always has exactly K actives -> both stats are the constant 32.0.
__global__ void sae_scalars(float* __restrict__ out)
{
    if (threadIdx.x == 0) { out[0] = 32.0f; out[1] = 32.0f; }
}

// ---------------------------------------------------------------------------
// Launch
// ---------------------------------------------------------------------------
extern "C" void kernel_launch(void* const* d_in, const int* in_sizes, int n_in,
                              void* d_out, int out_size, void* d_ws, size_t ws_size,
                              hipStream_t stream)
{
    (void)in_sizes; (void)n_in; (void)out_size; (void)d_ws; (void)ws_size;

    const int da = 768, db = 1024;

    const float* x_a     = (const float*)d_in[0];
    const float* x_b     = (const float*)d_in[1];
    const float* W_enc_a = (const float*)d_in[2];
    const float* W_dec_a = (const float*)d_in[3];
    const float* pre_b_a = (const float*)d_in[4];
    const float* lat_b_a = (const float*)d_in[5];
    const float* W_enc_b = (const float*)d_in[6];
    const float* W_dec_b = (const float*)d_in[7];
    const float* pre_b_b = (const float*)d_in[8];
    const float* lat_b_b = (const float*)d_in[9];

    float* out = (float*)d_out;

    // Flat output offsets in reference return order.
    const size_t oLogitsA = 0;
    const size_t oCodesA  = oLogitsA + (size_t)B_DIM * L_DIM;
    const size_t oMaskA   = oCodesA  + (size_t)B_DIM * L_DIM;
    const size_t oIdxA    = oMaskA   + (size_t)B_DIM * L_DIM;
    const size_t oLogitsB = oIdxA    + (size_t)B_DIM * K_TOP;
    const size_t oCodesB  = oLogitsB + (size_t)B_DIM * L_DIM;
    const size_t oMaskB   = oCodesB  + (size_t)B_DIM * L_DIM;
    const size_t oIdxB    = oMaskB   + (size_t)B_DIM * L_DIM;
    const size_t oReconA  = oIdxB    + (size_t)B_DIM * K_TOP;
    const size_t oReconB  = oReconA  + (size_t)B_DIM * da;
    const size_t oCrossBA = oReconB  + (size_t)B_DIM * db;
    const size_t oCrossAB = oCrossBA + (size_t)B_DIM * db;
    const size_t oStats   = oCrossAB + (size_t)B_DIM * da;

    const dim3 gemmGrid(L_DIM / BN, B_DIM / BM);   // (256, 32)
    const dim3 blk(256);

    // Encoders (bf16 WMMA).
    sae_encode_gemm<<<gemmGrid, blk, 0, stream>>>(x_a, W_enc_a, pre_b_a, lat_b_a,
                                                  out + oLogitsA, da);
    sae_encode_gemm<<<gemmGrid, blk, 0, stream>>>(x_b, W_enc_b, pre_b_b, lat_b_b,
                                                  out + oLogitsB, db);

    // Top-K gates (TDM row staging).
    sae_topk_gate<<<dim3(B_DIM), blk, 0, stream>>>(out + oLogitsA, out + oCodesA,
                                                   out + oMaskA, out + oIdxA);
    sae_topk_gate<<<dim3(B_DIM), blk, 0, stream>>>(out + oLogitsB, out + oCodesB,
                                                   out + oMaskB, out + oIdxB);

    // Sparse reconstructions (self + cross).
    sae_decode<<<dim3(B_DIM), blk, 0, stream>>>(out + oCodesA, out + oIdxA,
                                                W_dec_a, pre_b_a, out + oReconA, da);
    sae_decode<<<dim3(B_DIM), blk, 0, stream>>>(out + oCodesB, out + oIdxB,
                                                W_dec_b, pre_b_b, out + oReconB, db);
    sae_decode<<<dim3(B_DIM), blk, 0, stream>>>(out + oCodesA, out + oIdxA,
                                                W_dec_b, pre_b_b, out + oCrossBA, db);
    sae_decode<<<dim3(B_DIM), blk, 0, stream>>>(out + oCodesB, out + oIdxB,
                                                W_dec_a, pre_b_a, out + oCrossAB, da);

    // Sparsity stats (constant K).
    sae_scalars<<<dim3(1), dim3(32), 0, stream>>>(out + oStats);
}
